// SwinBlock2D_89034672046686
// MI455X (gfx1250) — compile-verified, matrix operand order checked by reading, not run on hardware
//
#include <hip/hip_runtime.h>
#include <hip/hip_bf16.h>

typedef _Float16 half_t;
typedef __attribute__((ext_vector_type(16))) _Float16 v16h;
typedef __attribute__((ext_vector_type(8)))  float    v8f;

#define LDH 392   // h16/o16 row stride (halves)
#define LDU 136   // mlp activation stride
#define LDQ 40    // q/k per-head stride
#define LDV 72    // vT stride
#define LDS_SC 72 // score stride

// f16 weight-fragment workspace offsets (in halves)
#define OFF_WQ 0
#define OFF_WK 147456
#define OFF_WV 294912
#define OFF_WO 442368
#define OFF_W1 589824
#define OFF_W2 1179648
#define TOTAL_DWORDS 884736  // total f16 pairs across all weights

// ---------------------------------------------------------------------------
// Prep: convert fp32 weights -> f16, pre-swizzled into WMMA B-fragment order.
// B tile = 32(K) x 16(N); lane L holds n = nt*16 + (L&15), half = L>>4;
// dword j<4 : k = kt*32 + half*8 + 2j ; j>=4 : k = kt*32 + 16 + half*8 + 2(j-4)
// ---------------------------------------------------------------------------
__global__ __launch_bounds__(256) void swin_prep(
    const float* __restrict__ wq, const float* __restrict__ wk,
    const float* __restrict__ wv, const float* __restrict__ wo,
    const float* __restrict__ w1, const float* __restrict__ w2,
    half_t* __restrict__ ws) {
  int g = blockIdx.x * blockDim.x + threadIdx.x;
  if (g >= TOTAL_DWORDS) return;
  const float* src; int N, Ntiles, local; long dsto;
  if (g < 294912) {                     // wq,wk,wv,wo (384x384 each)
    int mi = g / 73728; local = g % 73728;
    src = (mi == 0) ? wq : (mi == 1) ? wk : (mi == 2) ? wv : wo;
    N = 384; Ntiles = 24; dsto = (long)mi * 147456;
  } else if (g < 589824) {              // w1 (384x1536)
    local = g - 294912; src = w1; N = 1536; Ntiles = 96; dsto = OFF_W1;
  } else {                              // w2 (1536x384)
    local = g - 589824; src = w2; N = 384; Ntiles = 24; dsto = OFF_W2;
  }
  int j    = local & 7;
  int lane = (local >> 3) & 31;
  int tn   = local >> 8;
  int nt = tn % Ntiles, kt = tn / Ntiles;
  int n  = nt * 16 + (lane & 15);
  int hf = (lane >> 4) * 8;
  int k  = kt * 32 + ((j < 4) ? (hf + 2 * j) : (16 + hf + 2 * (j - 4)));
  ws[dsto + (long)local * 2]     = (half_t)src[(long)k * N + n];
  ws[dsto + (long)local * 2 + 1] = (half_t)src[(long)(k + 1) * N + n];
}

// ---------------------------------------------------------------------------
// Fragment helpers
// ---------------------------------------------------------------------------
// Row-major f16 LDS fragment: works for A (rows=m) and for B when the source
// is stored transposed so that B[k][n] = buf[n][k].
__device__ __forceinline__ v16h ld_frag(const half_t* p, int ld, int r0, int c0,
                                        int lane) {
  const half_t* rp = p + (r0 + (lane & 15)) * ld + c0 + ((lane >> 4) << 3);
  v16h f;
#pragma unroll
  for (int j = 0; j < 8; ++j) { f[j] = rp[j]; f[8 + j] = rp[16 + j]; }
  return f;
}
// Pre-swizzled weight fragment: one contiguous 32B load per lane.
__device__ __forceinline__ v16h ld_frag_w(const half_t* wbase, int Ntiles,
                                          int kt, int nt, int lane) {
  return *(const v16h*)(wbase + (((kt * Ntiles + nt) * 32 + lane) << 4));
}
__device__ __forceinline__ v8f wmma_f16(v16h a, v16h b, v8f c) {
  return __builtin_amdgcn_wmma_f32_16x16x32_f16(false, a, false, b, (short)0, c,
                                                false, false);
}
__device__ __forceinline__ float gelu_f(float x) {
  float t = tanhf(0.7978845608028654f * (x + 0.044715f * x * x * x));
  return 0.5f * x * (1.0f + t);
}

// ---------------------------------------------------------------------------
// Fused Swin block: one workgroup (8 wave32) per 7x7 window.
// ---------------------------------------------------------------------------
__global__ __launch_bounds__(256) void swin_block(
    const float* __restrict__ x, const float* __restrict__ ln1_s,
    const float* __restrict__ ln1_b, const float* __restrict__ bq,
    const float* __restrict__ bk, const float* __restrict__ bv,
    const float* __restrict__ bo, const float* __restrict__ ln2_s,
    const float* __restrict__ ln2_b, const float* __restrict__ b1,
    const float* __restrict__ b2, const half_t* __restrict__ ws,
    float* __restrict__ out) {
  extern __shared__ char smem[];
  float*  t_res = (float*)smem;          // [49][384] fp32 residual
  float*  row_m = t_res + 49 * 384;      // [64]
  float*  row_r = row_m + 64;            // [64]
  float*  s32   = row_r + 64;            // [64][72] raw scores
  half_t* h16   = (half_t*)(s32 + 64 * LDS_SC);  // [64][392]
  half_t* o16   = h16 + 64 * LDH;                // [64][392]
  half_t* u16   = o16 + 64 * LDH;                // [64][136]
  half_t* qh    = u16 + 64 * LDU;                // [64][40]
  half_t* kh    = qh + 64 * LDQ;                 // [64][40]
  half_t* vT    = kh + 64 * LDQ;                 // [32][72] (transposed V)
  half_t* a16   = vT + 32 * LDV;                 // [64][72] softmaxed attn

  const int tid  = threadIdx.x;
  const int lane = tid & 31;
  const int wid  = tid >> 5;
  const int ln15 = lane & 15;
  const int hf8  = (lane >> 4) << 3;  // 8*(lane>>4): D-tile row offset

  const int w  = blockIdx.x;
  const int b  = w >> 6;
  const int wy = (w >> 3) & 7;
  const int wx = w & 7;

  // ---- 1. gather shifted window -> fp32 residual ----
  for (int idx = tid; idx < 49 * 384; idx += 256) {
    int s = idx % 49, c = idx / 49;
    int gy = s / 7, gx = s % 7;
    int row = (wy * 7 + gy + 3) % 56;
    int col = (wx * 7 + gx + 3) % 56;
    t_res[s * 384 + c] = x[(((long)b * 384 + c) * 56 + row) * 56 + col];
  }
  for (int idx = tid; idx < 15 * LDH; idx += 256)  // zero M-pad rows of h
    h16[49 * LDH + idx] = (half_t)0.f;
  __syncthreads();

  // ---- 2. LN1 -> h16 ----
  if (tid < 49) {
    float sm = 0.f, sq = 0.f;
    for (int c = 0; c < 384; ++c) {
      float v = t_res[tid * 384 + c]; sm += v; sq += v * v;
    }
    float m = sm * (1.f / 384.f);
    row_m[tid] = m;
    row_r[tid] = rsqrtf(sq * (1.f / 384.f) - m * m + 1e-6f);
  }
  __syncthreads();
  for (int idx = tid; idx < 49 * 384; idx += 256) {
    int s = idx / 384, c = idx % 384;
    h16[s * LDH + c] =
        (half_t)((t_res[idx] - row_m[s]) * row_r[s] * ln1_s[c] + ln1_b[c]);
  }
  __syncthreads();

  v8f zero = {};

  // ---- 3. attention, head by head ----
#pragma unroll 1
  for (int hd = 0; hd < 12; ++hd) {
    const int mt = wid >> 1, nt = wid & 1;   // 8 tiles (4x2) -> 1 per wave
    const int ntg = hd * 2 + nt;
    {  // Q
      v8f acc = zero;
      for (int kt = 0; kt < 12; ++kt)
        acc = wmma_f16(ld_frag(h16, LDH, mt * 16, kt * 32, lane),
                       ld_frag_w(ws + OFF_WQ, 24, kt, ntg, lane), acc);
      float bb = bq[hd * 32 + nt * 16 + ln15];
#pragma unroll
      for (int v = 0; v < 8; ++v)
        qh[(mt * 16 + v + hf8) * LDQ + nt * 16 + ln15] = (half_t)(acc[v] + bb);
    }
    {  // K
      v8f acc = zero;
      for (int kt = 0; kt < 12; ++kt)
        acc = wmma_f16(ld_frag(h16, LDH, mt * 16, kt * 32, lane),
                       ld_frag_w(ws + OFF_WK, 24, kt, ntg, lane), acc);
      float bb = bk[hd * 32 + nt * 16 + ln15];
#pragma unroll
      for (int v = 0; v < 8; ++v)
        kh[(mt * 16 + v + hf8) * LDQ + nt * 16 + ln15] = (half_t)(acc[v] + bb);
    }
    {  // V, stored transposed: vT[dim][token]
      v8f acc = zero;
      for (int kt = 0; kt < 12; ++kt)
        acc = wmma_f16(ld_frag(h16, LDH, mt * 16, kt * 32, lane),
                       ld_frag_w(ws + OFF_WV, 24, kt, ntg, lane), acc);
      float bb = bv[hd * 32 + nt * 16 + ln15];
#pragma unroll
      for (int v = 0; v < 8; ++v)
        vT[(nt * 16 + ln15) * LDV + mt * 16 + hf8 + v] = (half_t)(acc[v] + bb);
    }
    __syncthreads();

    // scores = Q K^T / sqrt(dh): 16 tiles (4x4) -> 2 per wave; K=32 = 1 step
#pragma unroll
    for (int i = 0; i < 2; ++i) {
      int t2 = wid * 2 + i, smt = t2 >> 2, snt = t2 & 3;
      v8f acc = wmma_f16(ld_frag(qh, LDQ, smt * 16, 0, lane),
                         ld_frag(kh, LDQ, snt * 16, 0, lane), zero);
#pragma unroll
      for (int v = 0; v < 8; ++v)
        s32[(smt * 16 + v + hf8) * LDS_SC + snt * 16 + ln15] =
            acc[v] * 0.17677669529663687f;
    }
    __syncthreads();

    // fp32 softmax over 49 valid columns
    if (tid < 49) {
      float mx = -1e30f;
      for (int c = 0; c < 49; ++c) mx = fmaxf(mx, s32[tid * LDS_SC + c]);
      float sum = 0.f;
      for (int c = 0; c < 49; ++c) sum += __expf(s32[tid * LDS_SC + c] - mx);
      float inv = 1.f / sum;
      for (int c = 0; c < 49; ++c)
        a16[tid * LDS_SC + c] = (half_t)(__expf(s32[tid * LDS_SC + c] - mx) * inv);
      for (int c = 49; c < 64; ++c) a16[tid * LDS_SC + c] = (half_t)0.f;
    } else if (tid < 64) {  // zero M-pad rows feeding next GEMM
      for (int c = 0; c < 64; ++c) a16[tid * LDS_SC + c] = (half_t)0.f;
    }
    __syncthreads();

    // o_h = attn @ V: 8 tiles -> 1 per wave; K=64 (2 steps)
    {
      v8f acc = zero;
#pragma unroll
      for (int kk = 0; kk < 2; ++kk)
        acc = wmma_f16(ld_frag(a16, LDS_SC, mt * 16, kk * 32, lane),
                       ld_frag(vT, LDV, nt * 16, kk * 32, lane), acc);
#pragma unroll
      for (int v = 0; v < 8; ++v)
        o16[(mt * 16 + v + hf8) * LDH + hd * 32 + nt * 16 + ln15] =
            (half_t)acc[v];
    }
    __syncthreads();
  }

  // ---- 4. out projection + residual (96 tiles -> 12 per wave) ----
#pragma unroll 1
  for (int i = 0; i < 12; ++i) {
    int t4 = wid * 12 + i, mt = t4 / 24, nt = t4 % 24;
    v8f acc = zero;
    for (int kt = 0; kt < 12; ++kt)
      acc = wmma_f16(ld_frag(o16, LDH, mt * 16, kt * 32, lane),
                     ld_frag_w(ws + OFF_WO, 24, kt, nt, lane), acc);
    float bb = bo[nt * 16 + ln15];
#pragma unroll
    for (int v = 0; v < 8; ++v) {
      int r = mt * 16 + v + hf8;
      if (r < 49) t_res[r * 384 + nt * 16 + ln15] += acc[v] + bb;
    }
  }
  __syncthreads();

  // ---- 5. LN2 -> h16 ----
  if (tid < 49) {
    float sm = 0.f, sq = 0.f;
    for (int c = 0; c < 384; ++c) {
      float v = t_res[tid * 384 + c]; sm += v; sq += v * v;
    }
    float m = sm * (1.f / 384.f);
    row_m[tid] = m;
    row_r[tid] = rsqrtf(sq * (1.f / 384.f) - m * m + 1e-6f);
  }
  __syncthreads();
  for (int idx = tid; idx < 49 * 384; idx += 256) {
    int s = idx / 384, c = idx % 384;
    h16[s * LDH + c] =
        (half_t)((t_res[idx] - row_m[s]) * row_r[s] * ln2_s[c] + ln2_b[c]);
  }
  __syncthreads();

  // ---- 6. MLP, hidden dim tiled in blocks of 128 ----
  v8f acc2[12];
#pragma unroll
  for (int i = 0; i < 12; ++i) acc2[i] = zero;
#pragma unroll 1
  for (int kb = 0; kb < 12; ++kb) {
    // u = gelu(h @ w1[:, kb*128 : kb*128+128] + b1): 32 tiles -> 4 per wave
#pragma unroll 1
    for (int i = 0; i < 4; ++i) {
      int t6 = wid * 4 + i, mt = t6 >> 3, nt = t6 & 7;
      v8f acc = zero;
      for (int kt = 0; kt < 12; ++kt)
        acc = wmma_f16(ld_frag(h16, LDH, mt * 16, kt * 32, lane),
                       ld_frag_w(ws + OFF_W1, 96, kt, kb * 8 + nt, lane), acc);
      float bb = b1[kb * 128 + nt * 16 + ln15];
#pragma unroll
      for (int v = 0; v < 8; ++v)
        u16[(mt * 16 + v + hf8) * LDU + nt * 16 + ln15] =
            (half_t)gelu_f(acc[v] + bb);
    }
    __syncthreads();
    // t += u @ w2[kb*128 : kb*128+128, :]  (register-resident accumulators)
#pragma unroll
    for (int i = 0; i < 12; ++i) {
      int t7 = wid * 12 + i, mt = t7 / 24, nt = t7 % 24;
#pragma unroll
      for (int kk = 0; kk < 4; ++kk)
        acc2[i] = wmma_f16(ld_frag(u16, LDU, mt * 16, kk * 32, lane),
                           ld_frag_w(ws + OFF_W2, 24, kb * 4 + kk, nt, lane),
                           acc2[i]);
    }
    __syncthreads();
  }
#pragma unroll
  for (int i = 0; i < 12; ++i) {
    int t7 = wid * 12 + i, mt = t7 / 24, nt = t7 % 24;
    float bb = b2[nt * 16 + ln15];
#pragma unroll
    for (int v = 0; v < 8; ++v) {
      int r = mt * 16 + v + hf8;
      if (r < 49) t_res[r * 384 + nt * 16 + ln15] += acc2[i][v] + bb;
    }
  }
  __syncthreads();

  // ---- 7. scatter back (inverse shift == same coords as gather) ----
  for (int idx = tid; idx < 49 * 384; idx += 256) {
    int s = idx % 49, c = idx / 49;
    int gy = s / 7, gx = s % 7;
    int row = (wy * 7 + gy + 3) % 56;
    int col = (wx * 7 + gx + 3) % 56;
    out[(((long)b * 384 + c) * 56 + row) * 56 + col] = t_res[s * 384 + c];
  }
}

// ---------------------------------------------------------------------------
extern "C" void kernel_launch(void* const* d_in, const int* in_sizes, int n_in,
                              void* d_out, int out_size, void* d_ws,
                              size_t ws_size, hipStream_t stream) {
  (void)in_sizes; (void)n_in; (void)out_size; (void)ws_size;
  const float* x     = (const float*)d_in[0];
  const float* ln1_s = (const float*)d_in[1];
  const float* ln1_b = (const float*)d_in[2];
  const float* wq    = (const float*)d_in[3];
  const float* bq    = (const float*)d_in[4];
  const float* wk    = (const float*)d_in[5];
  const float* bk    = (const float*)d_in[6];
  const float* wv    = (const float*)d_in[7];
  const float* bv    = (const float*)d_in[8];
  const float* wo    = (const float*)d_in[9];
  const float* bo    = (const float*)d_in[10];
  const float* ln2_s = (const float*)d_in[11];
  const float* ln2_b = (const float*)d_in[12];
  const float* w1    = (const float*)d_in[13];
  const float* b1    = (const float*)d_in[14];
  const float* w2    = (const float*)d_in[15];
  const float* b2    = (const float*)d_in[16];
  half_t* ws = (half_t*)d_ws;
  float* out = (float*)d_out;

  swin_prep<<<(TOTAL_DWORDS + 255) / 256, 256, 0, stream>>>(wq, wk, wv, wo, w1,
                                                            w2, ws);

  // dynamic LDS: fp32 region + f16 region
  size_t smem = (size_t)(49 * 384 + 64 + 64 + 64 * LDS_SC) * 4 +
                (size_t)(64 * LDH * 2 + 64 * LDU + 64 * LDQ * 2 + 32 * LDV +
                         64 * LDS_SC) * 2;  // = 236,032 B
  swin_block<<<2048, 256, smem, stream>>>(x, ln1_s, ln1_b, bq, bk, bv, bo,
                                          ln2_s, ln2_b, b1, b2, ws, out);
}
